// QJLTransform_68032281969137
// MI455X (gfx1250) — matrix operand order repaired and checked by reading, not compiled, not run on hardware
//
#include <hip/hip_runtime.h>
#include <hip/hip_bf16.h>
#include <math.h>

typedef __attribute__((ext_vector_type(16))) _Float16 v16h;
typedef __attribute__((ext_vector_type(8)))  _Float16 v8h;
typedef __attribute__((ext_vector_type(4)))  _Float16 v4h;
typedef __attribute__((ext_vector_type(8)))  float    v8f;

#define QJL_BH 32      // B*H
#define QJL_SQ 2048
#define QJL_SK 2048
#define QJL_D  128
#define QJL_M  128
#define LDST   136     // padded LDS row stride (halfs): 272B = 16B aligned, conflict-free

// ---------------------------------------------------------------------------
// WMMA fragment loaders (per CDNA5 ISA 7.12.2 VGPR layouts, wave32)
// ---------------------------------------------------------------------------

// A fragment 16x32 f16: lane&15 = row; lanes<16 hold K {0..7,16..23}, lanes>=16 hold K {8..15,24..31}
__device__ __forceinline__ v16h load_frag_a(const _Float16* t, int row0, int k0, int lane) {
    const _Float16* p = t + (size_t)(row0 + (lane & 15)) * LDST + k0 + ((lane >> 4) << 3);
    v8h lo = *(const v8h*)(p);
    v8h hi = *(const v8h*)(p + 16);
    v16h r;
#pragma unroll
    for (int i = 0; i < 8; ++i) { r[i] = lo[i]; r[i + 8] = hi[i]; }
    return r;
}

// B fragment 32x16 f16: lane&15 = col; lanes<16 hold K {0..7,16..23}, lanes>=16 hold K {8..15,24..31}.
// LDS tile stored as B^T row-major: [col][k].
__device__ __forceinline__ v16h load_frag_b(const _Float16* t, int col0, int k0, int lane) {
    const _Float16* p = t + (size_t)(col0 + (lane & 15)) * LDST + k0 + ((lane >> 4) << 3);
    v8h lo = *(const v8h*)(p);
    v8h hi = *(const v8h*)(p + 16);
    v16h r;
#pragma unroll
    for (int i = 0; i < 8; ++i) { r[i] = lo[i]; r[i + 8] = hi[i]; }
    return r;
}

__device__ __forceinline__ v8f zero_v8f() {
    v8f z;
#pragma unroll
    for (int e = 0; e < 8; ++e) z[e] = 0.0f;
    return z;
}

// Core: each wave computes a 32x64 block of a 128x128 output tile; K = 128 (4 WMMA steps).
// Wave grid: 4 wave-rows x 2 wave-cols (8 waves / 256 threads).
__device__ __forceinline__ void wmma_block_128(const _Float16* ldsA, const _Float16* ldsB,
                                               v8f acc[2][4], int wr, int wc, int lane) {
#pragma unroll
    for (int kk = 0; kk < 128; kk += 32) {
        v16h a[2], b[4];
#pragma unroll
        for (int i = 0; i < 2; ++i) a[i] = load_frag_a(ldsA, wr * 32 + i * 16, kk, lane);
#pragma unroll
        for (int j = 0; j < 4; ++j) b[j] = load_frag_b(ldsB, wc * 64 + j * 16, kk, lane);
#pragma unroll
        for (int i = 0; i < 2; ++i)
#pragma unroll
            for (int j = 0; j < 4; ++j)
                acc[i][j] = __builtin_amdgcn_wmma_f32_16x16x32_f16(
                    false, a[i], false, b[j], (short)0, acc[i][j], false, false);
    }
}

// ---------------------------------------------------------------------------
// Kernel 1: per-row L2 norm of k. One wave32 per row of 128 floats.
// ---------------------------------------------------------------------------
__global__ __launch_bounds__(256) void qjl_norm_kernel(const float* __restrict__ k,
                                                       float* __restrict__ norms) {
    const int row  = blockIdx.x * 8 + (threadIdx.x >> 5);   // 8 waves per block
    const int lane = threadIdx.x & 31;
    const float4 v = *(const float4*)(k + (size_t)row * QJL_D + lane * 4);
    float s = v.x * v.x + v.y * v.y + v.z * v.z + v.w * v.w;
#pragma unroll
    for (int m = 16; m >= 1; m >>= 1) s += __shfl_xor(s, m, 32);
    if (lane == 0) norms[row] = sqrtf(s);
}

// ---------------------------------------------------------------------------
// Kernel 2: projection GEMM  Out[bh,row,m] = X[bh,row,:] . S[m,:]
//   IS_K=false -> store f16(proj)   (Sq)
//   IS_K=true  -> store sign(proj) as f16 +/-1
// Block: 128 rows x 128 m, f32->f16 conversion during LDS staging.
// ---------------------------------------------------------------------------
template <bool IS_K>
__global__ __launch_bounds__(256) void qjl_proj_kernel(const float* __restrict__ X,
                                                       const float* __restrict__ S,
                                                       _Float16* __restrict__ Out) {
    __shared__ _Float16 ldsX[128 * LDST];   // A tile: [row][d]
    __shared__ _Float16 ldsS[128 * LDST];   // B^T tile: [m][d]
    const int tid = threadIdx.x;
    const int rb  = blockIdx.x;
    const int bh  = blockIdx.y;

    const float* Xg = X + ((size_t)bh * QJL_SQ + (size_t)rb * 128) * QJL_D;
#pragma unroll 4
    for (int it = 0; it < 16; ++it) {                 // 4096 float4 chunks / 256 threads
        int c   = tid + it * 256;
        int row = c >> 5;
        int col = (c & 31) << 2;                      // float index within row
        float4 s4 = *(const float4*)(S  + row * QJL_D + col);
        float4 x4 = *(const float4*)(Xg + row * QJL_D + col);
        v4h sh; sh[0] = (_Float16)s4.x; sh[1] = (_Float16)s4.y; sh[2] = (_Float16)s4.z; sh[3] = (_Float16)s4.w;
        v4h xh; xh[0] = (_Float16)x4.x; xh[1] = (_Float16)x4.y; xh[2] = (_Float16)x4.z; xh[3] = (_Float16)x4.w;
        *(v4h*)(ldsS + (size_t)row * LDST + col) = sh;
        *(v4h*)(ldsX + (size_t)row * LDST + col) = xh;
    }
    __syncthreads();

    const int lane = tid & 31, wave = tid >> 5;
    const int wr = wave >> 1, wc = wave & 1;
    v8f acc[2][4];
#pragma unroll
    for (int i = 0; i < 2; ++i)
#pragma unroll
        for (int j = 0; j < 4; ++j) acc[i][j] = zero_v8f();

    wmma_block_128(ldsX, ldsS, acc, wr, wc, lane);

    _Float16* og = Out + ((size_t)bh * QJL_SQ + (size_t)rb * 128) * QJL_M;
#pragma unroll
    for (int i = 0; i < 2; ++i)
#pragma unroll
        for (int j = 0; j < 4; ++j) {
            const int col = wc * 64 + j * 16 + (lane & 15);
#pragma unroll
            for (int r = 0; r < 8; ++r) {
                const int row = wr * 32 + i * 16 + r + ((lane >> 4) << 3);
                const float v = acc[i][j][r];
                og[(size_t)row * QJL_M + col] =
                    IS_K ? (_Float16)((v >= 0.0f) ? 1.0f : -1.0f) : (_Float16)v;
            }
        }
}

// ---------------------------------------------------------------------------
// Kernel 3: main GEMM  out[bh,q,k] = (Sq[bh,q,:] . signs[bh,k,:]) * scale * norm[bh,k]
// Block: 128 q x 128 k per workgroup; K=128 staged once in LDS.
// ---------------------------------------------------------------------------
__global__ __launch_bounds__(256) void qjl_main_kernel(const _Float16* __restrict__ Sq,
                                                       const _Float16* __restrict__ Sg,
                                                       const float* __restrict__ norms,
                                                       float* __restrict__ out, float scale) {
    __shared__ _Float16 ldsA[128 * LDST];
    __shared__ _Float16 ldsB[128 * LDST];
    const int tid = threadIdx.x;
    const int qb = blockIdx.x, kb = blockIdx.y, bh = blockIdx.z;

    const _Float16* Ag = Sq + ((size_t)bh * QJL_SQ + (size_t)qb * 128) * QJL_M;
    const _Float16* Bg = Sg + ((size_t)bh * QJL_SK + (size_t)kb * 128) * QJL_M;
#pragma unroll 4
    for (int it = 0; it < 8; ++it) {                  // 2048 16B chunks / 256 threads
        int c   = tid + it * 256;
        int row = c >> 4;
        int col = (c & 15) << 3;                      // half index within row
        *(uint4*)(ldsA + (size_t)row * LDST + col) = *(const uint4*)(Ag + (size_t)row * QJL_M + col);
        *(uint4*)(ldsB + (size_t)row * LDST + col) = *(const uint4*)(Bg + (size_t)row * QJL_M + col);
    }
    __syncthreads();

    const int lane = tid & 31, wave = tid >> 5;
    const int wr = wave >> 1, wc = wave & 1;
    v8f acc[2][4];
#pragma unroll
    for (int i = 0; i < 2; ++i)
#pragma unroll
        for (int j = 0; j < 4; ++j) acc[i][j] = zero_v8f();

    wmma_block_128(ldsA, ldsB, acc, wr, wc, lane);

    const float* nrm = norms + (size_t)bh * QJL_SK + (size_t)kb * 128;
    float* og = out + ((size_t)bh * QJL_SQ + (size_t)qb * 128) * QJL_SK + (size_t)kb * 128;
#pragma unroll
    for (int i = 0; i < 2; ++i)
#pragma unroll
        for (int j = 0; j < 4; ++j) {
            const int col = wc * 64 + j * 16 + (lane & 15);
            const float nv = nrm[col] * scale;
#pragma unroll
            for (int r = 0; r < 8; ++r) {
                const int row = wr * 32 + i * 16 + r + ((lane >> 4) << 3);
                og[(size_t)row * QJL_SK + col] = acc[i][j][r] * nv;
            }
        }
}

// ---------------------------------------------------------------------------
extern "C" void kernel_launch(void* const* d_in, const int* in_sizes, int n_in,
                              void* d_out, int out_size, void* d_ws, size_t ws_size,
                              hipStream_t stream) {
    const float* q = (const float*)d_in[0];   // (B,H,SQ,D) f32
    const float* k = (const float*)d_in[1];   // (B,H,SK,D) f32
    const float* S = (const float*)d_in[2];   // (M,D)      f32
    float* out = (float*)d_out;               // (B,H,SQ,SK) f32

    // workspace: Sq_f16 (16.8MB) | signs_f16 (16.8MB) | norms (256KB)
    _Float16* Sq    = (_Float16*)d_ws;
    _Float16* Signs = Sq + (size_t)QJL_BH * QJL_SQ * QJL_M;
    float*    norms = (float*)(Signs + (size_t)QJL_BH * QJL_SK * QJL_M);

    const float scale = 1.2533141373155003f / (float)QJL_M;   // sqrt(pi/2)/M

    qjl_norm_kernel<<<dim3((QJL_BH * QJL_SK) / 8), 256, 0, stream>>>(k, norms);
    qjl_proj_kernel<false><<<dim3(QJL_SQ / 128, QJL_BH), 256, 0, stream>>>(q, S, Sq);
    qjl_proj_kernel<true ><<<dim3(QJL_SK / 128, QJL_BH), 256, 0, stream>>>(k, S, Signs);
    qjl_main_kernel<<<dim3(QJL_SQ / 128, QJL_SK / 128, QJL_BH), 256, 0, stream>>>(
        Sq, Signs, norms, out, scale);
}
